// MambaBlock_63367947485692
// MI455X (gfx1250) — matrix operand order repaired and checked by reading, not compiled, not run on hardware
//
#include <hip/hip_runtime.h>

// ---------------------------------------------------------------------------
// Mamba block forward, MI455X (gfx1250, wave32)
// bf16 WMMA GEMMs with double-buffered global_load_async_to_lds_b128 staging.
// B=2, L=2048, H=1024, E=2048, K=4, N=16, DT_RANK=128
// ---------------------------------------------------------------------------

typedef __attribute__((ext_vector_type(16))) __bf16 v16bf;
typedef __attribute__((ext_vector_type(8)))  float  v8f;

union Frag16 {            // 16 bf16 = 8 dwords per lane
    unsigned int u[8];
    v16bf        v;
};

__device__ __forceinline__ unsigned short f2bf(float f) {
    unsigned int u = __float_as_uint(f);
    u += 0x7FFFu + ((u >> 16) & 1u);        // round-to-nearest-even
    return (unsigned short)(u >> 16);
}
__device__ __forceinline__ float siluf(float x) { return x / (1.0f + __expf(-x)); }
__device__ __forceinline__ float softplusf(float x) {
    return (x > 20.0f) ? x : __logf(1.0f + __expf(x));
}

// raw LDS byte offset of a __shared__ object (generic -> addrspace(3) -> int)
__device__ __forceinline__ unsigned ldsOffset(const void* p) {
    return (unsigned)(unsigned long long)
           (__attribute__((address_space(3))) const void*)p;
}

// A-fragment: 16x32 bf16 (this code's contiguous k = kh + s, same as B): lane m=lane&15, kh=(lane>>4)*16
__device__ __forceinline__ void loadAfrag(const unsigned short* As, int mBase,
                                          int lane, Frag16& fr) {
    const unsigned short* base = As + (mBase + (lane & 15)) * 32;
    int kh = (lane >> 4) * 16;
#pragma unroll
    for (int v = 0; v < 4; ++v)
        fr.u[v] = *(const unsigned int*)(base + kh + 2 * v);
#pragma unroll
    for (int v = 0; v < 4; ++v)
        fr.u[4 + v] = *(const unsigned int*)(base + 8 + kh + 2 * v);
}

// B-fragment: 32x16 bf16, LDS tile is n-major [n][k] so K-pairs contiguous
__device__ __forceinline__ void loadBfrag(const unsigned short* Bs, int nBase,
                                          int lane, Frag16& fr) {
    const unsigned short* base = Bs + (nBase + (lane & 15)) * 32 + (lane >> 4) * 16;
#pragma unroll
    for (int v = 0; v < 8; ++v)
        fr.u[v] = *(const unsigned int*)(base + 2 * v);
}

// ---------------------------------------------------------------------------
// WMMA GEMM: C[M,N] = A[M,K] (bf16, row-major) * Bt[N,K]^T (bf16, n-major)
// 128x128 tile, K-step 32, 8 waves, each wave 4(M) x 2(N) WMMA tiles.
// Tiles staged with async DMA to LDS, double buffered.
// EP: 0 = +bias -> out0(f32)
//     1 = +bias, split: cols<N/2 -> out0(f32), silu -> out1f(f32)
//     2 = +bias, softplus -> out0(f32)
//     4 = plain -> out0(f32, ld0) and cols<128 -> bf16 out1b (ld 128)
// ---------------------------------------------------------------------------
template <int EP>
__global__ __launch_bounds__(256) void wmma_gemm(
    const unsigned short* __restrict__ A,
    const unsigned short* __restrict__ Bt,
    float* __restrict__ out0, float* __restrict__ out1f,
    unsigned short* __restrict__ out1b,
    const float* __restrict__ bias,
    int M, int N, int Kdim, int lda, int ldb, int ld0)
{
    __shared__ __align__(16) unsigned short As[2][128 * 32];
    __shared__ __align__(16) unsigned short Bs[2][128 * 32];

    const int tid  = threadIdx.x;
    const int lane = tid & 31;
    const int wid  = tid >> 5;
    const int waveM = wid & 1;        // 64 rows each
    const int waveN = wid >> 1;       // 32 cols each
    const int mBase = blockIdx.y * 128;
    const int nBase = blockIdx.x * 128;

    const unsigned asOff0 = ldsOffset(&As[0][0]);
    const unsigned asOff1 = ldsOffset(&As[1][0]);
    const unsigned bsOff0 = ldsOffset(&Bs[0][0]);
    const unsigned bsOff1 = ldsOffset(&Bs[1][0]);

    // each lane copies 2 x 16B chunks per matrix per tile (128 rows x 64B)
    auto issueTile = [&](unsigned ldsB, const unsigned short* src, int rowBase,
                         int ld, int rowMax, int k0) {
#pragma unroll
        for (int i = 0; i < 2; ++i) {
            int c    = tid + i * 256;          // 0..511
            int row  = c >> 2;                 // 0..127
            int col8 = (c & 3) * 8;            // bf16 element offset
            int gR   = rowBase + row;
            gR = (gR < rowMax) ? gR : (rowMax - 1);   // clamp (masked later)
            unsigned long long ga =
                (unsigned long long)(src + (size_t)gR * ld + k0 + col8);
            unsigned ldsA = ldsB + (unsigned)(row * 64 + col8 * 2);
            asm volatile("global_load_async_to_lds_b128 %0, %1, off"
                         :: "v"(ldsA), "v"(ga) : "memory");
        }
    };

    const int nT = Kdim >> 5;
    issueTile(asOff0, A,  mBase, lda, M, 0);
    issueTile(bsOff0, Bt, nBase, ldb, N, 0);

    v8f acc[4][2] = {};

    for (int t = 0; t < nT; ++t) {
        const int cur = t & 1;
        __syncthreads();                       // prev compute done -> nxt buf free
        if (t + 1 < nT) {
            unsigned aN = cur ? asOff0 : asOff1;
            unsigned bN = cur ? bsOff0 : bsOff1;
            issueTile(aN, A,  mBase, lda, M, (t + 1) * 32);
            issueTile(bN, Bt, nBase, ldb, N, (t + 1) * 32);
            asm volatile("s_wait_asynccnt 0x4" ::: "memory");  // cur tile done
        } else {
            asm volatile("s_wait_asynccnt 0x0" ::: "memory");
        }
        __syncthreads();                       // all waves' cur copies visible

        const unsigned short* Ac = &As[cur][0];
        const unsigned short* Bc = &Bs[cur][0];
        Frag16 af[4], bf[2];
#pragma unroll
        for (int mi = 0; mi < 4; ++mi)
            loadAfrag(Ac, waveM * 64 + mi * 16, lane, af[mi]);
#pragma unroll
        for (int ni = 0; ni < 2; ++ni)
            loadBfrag(Bc, waveN * 32 + ni * 16, lane, bf[ni]);
#pragma unroll
        for (int mi = 0; mi < 4; ++mi)
#pragma unroll
            for (int ni = 0; ni < 2; ++ni)
                acc[mi][ni] = __builtin_amdgcn_wmma_f32_16x16x32_bf16(
                    false, af[mi].v, false, bf[ni].v,
                    (short)0, acc[mi][ni], false, false);
    }

    // epilogue: C layout M = r + 8*(lane>=16), N = lane&15
    const int colLane = lane & 15;
    const int rowOff  = (lane >> 4) * 8;
#pragma unroll
    for (int mi = 0; mi < 4; ++mi) {
#pragma unroll
        for (int ni = 0; ni < 2; ++ni) {
            int col = nBase + waveN * 32 + ni * 16 + colLane;
            if (col >= N) continue;
#pragma unroll
            for (int r = 0; r < 8; ++r) {
                int row = mBase + waveM * 64 + mi * 16 + r + rowOff;
                float v = acc[mi][ni][r];
                if (EP == 0) {
                    out0[(size_t)row * ld0 + col] = v + bias[col];
                } else if (EP == 1) {
                    v += bias[col];
                    int Eh = N >> 1;
                    if (col < Eh) out0[(size_t)row * ld0 + col] = v;
                    else          out1f[(size_t)row * ld0 + (col - Eh)] = siluf(v);
                } else if (EP == 2) {
                    out0[(size_t)row * ld0 + col] = softplusf(v + bias[col]);
                } else {                      // EP == 4
                    out0[(size_t)row * ld0 + col] = v;
                    if (col < 128) out1b[(size_t)row * 128 + col] = f2bf(v);
                }
            }
        }
    }
}

// ---------------------------------------------------------------------------
// f32 -> bf16 flat convert
// ---------------------------------------------------------------------------
__global__ __launch_bounds__(256) void cvt_bf16_flat(
    const float* __restrict__ s, unsigned short* __restrict__ d, int n)
{
    int i = blockIdx.x * 256 + threadIdx.x;
    if (i < n) d[i] = f2bf(s[i]);
}

// ---------------------------------------------------------------------------
// f32 [R][C] -> bf16 transposed [C][R] (LDS tile transpose)
// ---------------------------------------------------------------------------
__global__ __launch_bounds__(256) void cvt_transpose_bf16(
    const float* __restrict__ src, unsigned short* __restrict__ dst, int R, int C)
{
    __shared__ unsigned short t[32][33];
    int bx = blockIdx.x * 32;     // col base (src)
    int by = blockIdx.y * 32;     // row base (src)
    int tx = threadIdx.x & 31, ty = threadIdx.x >> 5;   // ty: 0..7
#pragma unroll
    for (int i = 0; i < 32; i += 8) {
        int r = by + ty + i, c = bx + tx;
        float v = (r < R && c < C) ? src[(size_t)r * C + c] : 0.f;
        t[ty + i][tx] = f2bf(v);
    }
    __syncthreads();
#pragma unroll
    for (int i = 0; i < 32; i += 8) {
        int r = bx + ty + i;      // dst row = src col
        int c = by + tx;          // dst col = src row
        if (r < C && c < R) dst[(size_t)r * R + c] = t[tx][ty + i];
    }
}

// ---------------------------------------------------------------------------
// Depthwise causal conv (K=4) + bias + silu; writes f32 (scan) + bf16 (GEMM)
// ---------------------------------------------------------------------------
__global__ __launch_bounds__(256) void dw_conv_silu(
    const float* __restrict__ xin, const float* __restrict__ w,
    const float* __restrict__ b, float* __restrict__ xout,
    unsigned short* __restrict__ xoutb)
{
    const int E = 2048, L = 2048;
    int idx = blockIdx.x * 256 + threadIdx.x;          // (b,l,e) flat
    int e = idx & (E - 1);
    int l = (idx >> 11) & (L - 1);
    int bb = idx >> 22;
    float acc = b[e];
#pragma unroll
    for (int j = 0; j < 4; ++j) {
        int li = l - 3 + j;
        if (li >= 0) acc += w[e * 4 + j] * xin[((size_t)bb * L + li) * E + e];
    }
    float v = siluf(acc);
    xout[(size_t)idx]  = v;
    xoutb[(size_t)idx] = f2bf(v);
}

// ---------------------------------------------------------------------------
// Selective scan: one thread per (b,e,n); y reduced across 16-lane group.
// Output written directly in bf16 for the out-projection GEMM.
// ---------------------------------------------------------------------------
__global__ __launch_bounds__(256) void mamba_scan(
    const float* __restrict__ dlt, const float* __restrict__ u_,
    const float* __restrict__ proj, const float* __restrict__ xr,
    const float* __restrict__ A_log, const float* __restrict__ Dp,
    unsigned short* __restrict__ yzb)
{
    const int E = 2048, L = 2048;
    int gid = blockIdx.x * 256 + threadIdx.x;          // 65536 = B*E*16
    int n = gid & 15;
    int e = (gid >> 4) & (E - 1);
    int b = gid >> 15;

    float Av = -__expf(A_log[e * 16 + n]);
    float Dv = Dp[e];
    float h  = 0.f;

    for (int t = 0; t < L; ++t) {
        size_t row = (size_t)b * L + t;
        float d  = dlt[row * E + e];
        float u  = u_[row * E + e];
        float bn = proj[row * 160 + 128 + n];
        float cn = proj[row * 160 + 144 + n];
        h = __expf(d * Av) * h + (d * u) * bn;
        float part = h * cn;
        part += __shfl_xor(part, 1, 32);
        part += __shfl_xor(part, 2, 32);
        part += __shfl_xor(part, 4, 32);
        part += __shfl_xor(part, 8, 32);
        if (n == 0)
            yzb[row * E + e] = f2bf((part + u * Dv) * xr[row * E + e]);
    }
}

// ---------------------------------------------------------------------------
extern "C" void kernel_launch(void* const* d_in, const int* in_sizes, int n_in,
                              void* d_out, int out_size, void* d_ws, size_t ws_size,
                              hipStream_t stream)
{
    const float* x        = (const float*)d_in[0];
    const float* W_in     = (const float*)d_in[1];
    const float* b_in     = (const float*)d_in[2];
    const float* conv_w   = (const float*)d_in[3];
    const float* conv_b   = (const float*)d_in[4];
    const float* x_proj_w = (const float*)d_in[5];
    const float* dt_projw = (const float*)d_in[6];
    const float* dt_bias  = (const float*)d_in[7];
    const float* A_log    = (const float*)d_in[8];
    const float* Dp       = (const float*)d_in[9];
    const float* W_out    = (const float*)d_in[10];
    const float* b_out    = (const float*)d_in[11];
    float* out = (float*)d_out;

    // ---- workspace carve (256B aligned) ----
    size_t cur = 0;
    auto take = [&](size_t bytes) -> char* {
        char* p = (char*)d_ws + cur;
        cur += (bytes + 255) & ~(size_t)255;
        return p;
    };
    const size_t ME = (size_t)4096 * 2048;             // big activation elems

    unsigned short* xbf   = (unsigned short*)take((size_t)4096 * 1024 * 2);
    unsigned short* WtIn  = (unsigned short*)take((size_t)4096 * 1024 * 2);
    unsigned short* WtXp  = (unsigned short*)take((size_t)160  * 2048 * 2);
    unsigned short* WtDt  = (unsigned short*)take((size_t)2048 * 128  * 2);
    unsigned short* WtOut = (unsigned short*)take((size_t)1024 * 2048 * 2);
    float* xc_pre  = (float*)take(ME * 4);             // reused for yzb below
    float* xr      = (float*)take(ME * 4);
    float* xcv     = (float*)take(ME * 4);
    unsigned short* xcvbf = (unsigned short*)take(ME * 2);
    float* proj    = (float*)take((size_t)4096 * 160 * 4);
    unsigned short* dtbf = (unsigned short*)take((size_t)4096 * 128 * 2);
    float* dlt     = (float*)take(ME * 4);
    unsigned short* yzb = (unsigned short*)xc_pre;     // overlay (dead after conv)

    dim3 blk(256);

    // ---- precision conversions (once per launch) ----
    cvt_bf16_flat<<<16384, blk, 0, stream>>>(x, xbf, 4096 * 1024);
    cvt_transpose_bf16<<<dim3(128, 32), blk, 0, stream>>>(W_in,     WtIn, 1024, 4096);
    cvt_transpose_bf16<<<dim3(5,   64), blk, 0, stream>>>(x_proj_w, WtXp, 2048, 160);
    cvt_transpose_bf16<<<dim3(64,   4), blk, 0, stream>>>(dt_projw, WtDt, 128,  2048);
    cvt_transpose_bf16<<<dim3(32,  64), blk, 0, stream>>>(W_out,    WtOut, 2048, 1024);

    // 1) z = x @ W_in + b_in ; split -> xc_pre / silu -> xr
    wmma_gemm<1><<<dim3(32, 32), blk, 0, stream>>>(
        xbf, WtIn, xc_pre, xr, nullptr, b_in, 4096, 4096, 1024, 1024, 1024, 2048);

    // 2) causal depthwise conv + bias + silu -> xcv (f32) + xcvbf (bf16)
    dw_conv_silu<<<32768, blk, 0, stream>>>(xc_pre, conv_w, conv_b, xcv, xcvbf);

    // 3) proj = xc @ x_proj_w  (N=160) -> proj f32 (+ dt cols as bf16)
    wmma_gemm<4><<<dim3(2, 32), blk, 0, stream>>>(
        xcvbf, WtXp, proj, nullptr, dtbf, nullptr, 4096, 160, 2048, 2048, 2048, 160);

    // 4) delta = softplus(dt @ dt_proj_w + dt_bias)
    wmma_gemm<2><<<dim3(16, 32), blk, 0, stream>>>(
        dtbf, WtDt, dlt, nullptr, nullptr, dt_bias, 4096, 2048, 128, 128, 128, 2048);

    // 5) selective scan + gate -> yzb (bf16)
    mamba_scan<<<256, blk, 0, stream>>>(dlt, xcv, proj, xr, A_log, Dp, yzb);

    // 6) out = yz @ W_out + b_out
    wmma_gemm<0><<<dim3(8, 32), blk, 0, stream>>>(
        yzb, WtOut, out, nullptr, nullptr, b_out, 4096, 1024, 2048, 2048, 2048, 1024);
}
